// BiLSTMCRF_72335839199698
// MI455X (gfx1250) — matrix operand order repaired and checked by reading, not compiled, not run on hardware
//
#include <hip/hip_runtime.h>
#include <hip/hip_bf16.h>
#include <math.h>

// Problem constants (from reference)
#define BB   64
#define TT   1024
#define DD   512
#define HH   256
#define G4   1024   // 4*H
#define KK   52     // TAGSET + 2
#define START_TAG 50
#define STOP_TAG  51

typedef __attribute__((ext_vector_type(16))) __bf16 v16bf;
typedef __attribute__((ext_vector_type(8)))  float  v8f;

union FragU { v16bf v; unsigned short u[16]; uint4 q[2]; };

// Native bf16 conversion (RNE) — backend picks v_cvt*_bf16_f32.
__device__ __forceinline__ unsigned short f2bf(float x) {
  __bf16 h = (__bf16)x;
  return __builtin_bit_cast(unsigned short, h);
}
__device__ __forceinline__ unsigned pack_bf16(float lo, float hi) {
  return (unsigned)f2bf(lo) | ((unsigned)f2bf(hi) << 16);
}
__device__ __forceinline__ float sigmf(float x) { return 1.0f / (1.0f + __expf(-x)); }
// Fast tanh: e = exp(-2|x|) in (0,1] (no overflow), sign restored.
__device__ __forceinline__ float tanh_fast(float x) {
  float a = __builtin_fabsf(x);
  float e = __expf(-2.0f * a);
  float t = (1.0f - e) / (1.0f + e);
  return __builtin_copysignf(t, x);
}

// A fragment: 16x32 (MxK) bf16. Lane l<16 holds row m0+l, k = {kk..kk+7, kk+16..kk+23};
// lane l>=16 holds row m0+(l&15), k = {kk+8..kk+15, kk+24..kk+31}.
__device__ __forceinline__ v16bf ldA(const unsigned short* __restrict__ base, int lda,
                                     int m0, int kk, int lane) {
  int m = m0 + (lane & 15);
  int c = (lane < 16) ? 0 : 8;
  const unsigned short* p = base + (size_t)m * lda + kk + c;
  FragU f;
  f.q[0] = *(const uint4*)p;         // k = c .. c+7
  f.q[1] = *(const uint4*)(p + 16);  // k = c+16 .. c+23
  return f.v;
}
// B fragment: 32x16 (KxN) bf16 with B[k][n] = Wrow[n][k] (row-major NxK source).
// Lane l<16 -> n = n0+l, k = {kk..kk+7, kk+16..kk+23}; lane l>=16 -> n = n0+(l&15), k = {kk+8..kk+15, kk+24..kk+31}.
__device__ __forceinline__ v16bf ldB(const unsigned short* __restrict__ base, int ldb,
                                     int n0, int kk, int lane) {
  int n = n0 + (lane & 15);
  int c = (lane < 16) ? 0 : 8;
  const unsigned short* p = base + (size_t)n * ldb + kk + c;
  FragU f;
  f.q[0] = *(const uint4*)p;         // k = c .. c+7
  f.q[1] = *(const uint4*)(p + 16);  // k = c+16 .. c+23
  return f.v;
}
#define WMMA_BF16(A, Bq, C) \
  __builtin_amdgcn_wmma_f32_16x16x32_bf16(false, (A), false, (Bq), (short)0, (C), false, false)

// ---------------------------------------------------------------------------
// K0: f32 -> bf16 conversion (Whh weights, reused every recurrence step)
// ---------------------------------------------------------------------------
__global__ void k_cvt_bf16(const float* __restrict__ src, unsigned short* __restrict__ dst, int n) {
  int i = blockIdx.x * blockDim.x + threadIdx.x;
  if (i < n) dst[i] = f2bf(src[i]);
}

// ---------------------------------------------------------------------------
// K1: xg = x @ Wih^T + bih + bhh for both directions.  C tile 64x64 per block,
// 4 waves each own a 32x32 quadrant (2x2 WMMA 16x16x32 subtiles), K staged in
// LDS as bf16 in 32-wide chunks.  x (134 MB) and Wih stay L2-resident, so
// cross-block rereads cost L2 bandwidth, not HBM.
// ---------------------------------------------------------------------------
__global__ void __launch_bounds__(128) k_xg_gemm(
    const float* __restrict__ x,
    const float* __restrict__ Wf, const float* __restrict__ b1f, const float* __restrict__ b2f,
    const float* __restrict__ Wb, const float* __restrict__ b1b, const float* __restrict__ b2b,
    float* __restrict__ xgf, float* __restrict__ xgb)
{
  const int dir = blockIdx.z;
  const float* W  = dir ? Wb  : Wf;
  const float* b1 = dir ? b1b : b1f;
  const float* b2 = dir ? b2b : b2f;
  float* xg       = dir ? xgb : xgf;

  const int m0 = blockIdx.x * 64;
  const int n0 = blockIdx.y * 64;

  __shared__ unsigned short As[64 * 40];  // 40-half row stride: 16B aligned, bank-spread
  __shared__ unsigned short Bs[64 * 40];

  const int tid = threadIdx.x;
  const int lane = tid & 31;
  const int wave = tid >> 5;
  const int wm = wave >> 1, wn = wave & 1;

  v8f acc[2][2] = {};

  for (int kk = 0; kk < DD; kk += 32) {
    // stage 64x32 tiles of x and W, converting f32 -> bf16 (native cvt)
#pragma unroll
    for (int i = 0; i < 4; ++i) {
      int idx = i * 128 + tid;       // 512 float4 slots
      int r = idx >> 3;
      int c4 = (idx & 7) * 4;
      float4 va = *(const float4*)&x[(size_t)(m0 + r) * DD + kk + c4];
      uint2 pa;
      pa.x = pack_bf16(va.x, va.y);
      pa.y = pack_bf16(va.z, va.w);
      *(uint2*)&As[r * 40 + c4] = pa;
      float4 vb = *(const float4*)&W[(size_t)(n0 + r) * DD + kk + c4];
      uint2 pb;
      pb.x = pack_bf16(vb.x, vb.y);
      pb.y = pack_bf16(vb.z, vb.w);
      *(uint2*)&Bs[r * 40 + c4] = pb;
    }
    __syncthreads();

    v16bf a0 = ldA(As, 40, wm * 32,      0, lane);
    v16bf a1 = ldA(As, 40, wm * 32 + 16, 0, lane);
    v16bf q0 = ldB(Bs, 40, wn * 32,      0, lane);
    v16bf q1 = ldB(Bs, 40, wn * 32 + 16, 0, lane);
    acc[0][0] = WMMA_BF16(a0, q0, acc[0][0]);
    acc[0][1] = WMMA_BF16(a0, q1, acc[0][1]);
    acc[1][0] = WMMA_BF16(a1, q0, acc[1][0]);
    acc[1][1] = WMMA_BF16(a1, q1, acc[1][1]);
    __syncthreads();
  }

#pragma unroll
  for (int ms = 0; ms < 2; ++ms)
#pragma unroll
    for (int ns = 0; ns < 2; ++ns) {
      int col = n0 + wn * 32 + ns * 16 + (lane & 15);
      float bias = b1[col] + b2[col];
#pragma unroll
      for (int r = 0; r < 8; ++r) {
        int row = m0 + wm * 32 + ms * 16 + r + ((lane < 16) ? 0 : 8);
        xg[(size_t)row * G4 + col] = acc[ms][ns][r] + bias;
      }
    }
}

// ---------------------------------------------------------------------------
// K2: LSTM recurrence (one block = one direction, 16 waves).  h kept in LDS as
// bf16 (A operand of next step) and mirrored in per-lane f32 registers for the
// masked-hold path; c and len kept entirely in registers.  Gates = xg[t] +
// h @ Whh^T via 128 WMMAs/wave/step; Whh streamed bf16 from L2.  Next step's
// xg slab (256 KB) is prefetched while this step's WMMAs run.
// Wave w owns H-columns [16w, 16w+16).
// ---------------------------------------------------------------------------
__global__ void __launch_bounds__(512) k_lstm(
    const float* __restrict__ xgf, const float* __restrict__ xgb,
    const unsigned short* __restrict__ whf, const unsigned short* __restrict__ whb,
    const int* __restrict__ lens,
    unsigned short* __restrict__ hs)
{
  const int dir = blockIdx.x;
  const float* xg = dir ? xgb : xgf;
  const unsigned short* Whh = dir ? whb : whf;
  const int hofs = dir * HH;

  __shared__ unsigned short hbuf[64 * 264];  // 264-half stride: 16B aligned, bank-spread

  const int tid = threadIdx.x;
  const int lane = tid & 31;
  const int wave = tid >> 5;   // 0..15
  const int hc0 = wave * 16;
  const int colL = lane & 15;
  const int roff = (lane < 16) ? 0 : 8;

  for (int i = tid; i < 64 * 264; i += 512) hbuf[i] = 0;
  __syncthreads();

  v8f cacc[4] = {};   // cell state, accumulator layout
  v8f hprev[4] = {};  // previous h (f32, own columns), accumulator layout
  int lenreg[4][8];   // per-lane sequence lengths (loop-invariant)
#pragma unroll
  for (int mt = 0; mt < 4; ++mt)
#pragma unroll
    for (int r = 0; r < 8; ++r) lenreg[mt][r] = lens[mt * 16 + r + roff];

  for (int tt = 0; tt < TT; ++tt) {
    const int t = dir ? (TT - 1 - tt) : tt;

    // accumulators seeded with xg[t] (gate pre-activations)
    v8f acc[4][4];
#pragma unroll
    for (int q = 0; q < 4; ++q)
#pragma unroll
      for (int mt = 0; mt < 4; ++mt)
#pragma unroll
        for (int r = 0; r < 8; ++r) {
          int b = mt * 16 + r + roff;
          acc[q][mt][r] = xg[((size_t)b * TT + t) * G4 + q * HH + hc0 + colL];
        }

    // prefetch next step's xg slab: 64 rows x 4KB = 4096 cache lines,
    // 8 lines per thread; overlaps with this step's WMMAs.
    if (tt + 1 < TT) {
      const int tn = dir ? (TT - 2 - tt) : (tt + 1);
#pragma unroll
      for (int k = 0; k < 8; ++k) {
        int L = tid + k * 512;          // 0..4095
        int row = L >> 6;               // batch row b
        int lin = L & 63;               // 64B line within row
        __builtin_prefetch(&xg[((size_t)row * TT + tn) * G4 + (size_t)lin * 16], 0, 1);
      }
    }

    // gates += h @ Whh^T
    for (int kk = 0; kk < HH; kk += 32) {
      v16bf a[4];
#pragma unroll
      for (int mt = 0; mt < 4; ++mt) a[mt] = ldA(hbuf, 264, mt * 16, kk, lane);
#pragma unroll
      for (int q = 0; q < 4; ++q) {
        v16bf bq = ldB(Whh, HH, q * HH + hc0, kk, lane);
#pragma unroll
        for (int mt = 0; mt < 4; ++mt) acc[q][mt] = WMMA_BF16(a[mt], bq, acc[q][mt]);
      }
    }

    // fused nonlinearity + masked state update (all register-resident)
    unsigned short hkeep_bf[4][8];
    unsigned short hout_bf[4][8];
#pragma unroll
    for (int mt = 0; mt < 4; ++mt)
#pragma unroll
      for (int r = 0; r < 8; ++r) {
        bool m = t < lenreg[mt][r];
        float iv = sigmf(acc[0][mt][r]);
        float fv = sigmf(acc[1][mt][r]);
        float gv = tanh_fast(acc[2][mt][r]);
        float ov = sigmf(acc[3][mt][r]);
        float cprev = cacc[mt][r];
        float cn = fv * cprev + iv * gv;
        float hn = ov * tanh_fast(cn);
        cacc[mt][r] = m ? cn : cprev;
        float hk = m ? hn : hprev[mt][r];        // recurrent state (held when masked)
        hprev[mt][r] = hk;
        hkeep_bf[mt][r] = f2bf(hk);
        hout_bf[mt][r]  = f2bf(m ? hn : 0.0f);   // hs output (zeroed by mask)
      }

    __syncthreads();  // all GEMM reads of hbuf complete before overwrite
#pragma unroll
    for (int mt = 0; mt < 4; ++mt)
#pragma unroll
      for (int r = 0; r < 8; ++r) {
        int b = mt * 16 + r + roff;
        hbuf[b * 264 + hc0 + colL] = hkeep_bf[mt][r];
        hs[((size_t)b * TT + t) * (2 * HH) + hofs + hc0 + colL] = hout_bf[mt][r];
      }
    __syncthreads();
  }
}

// ---------------------------------------------------------------------------
// K3: emissions = [hf|hb] @ W_tag^T + b_tag.  A (bf16 states) loaded straight
// from global; W_tag staged bf16 in LDS (rows 52..63 zero-padded).
// ---------------------------------------------------------------------------
__global__ void __launch_bounds__(128) k_emit(
    const unsigned short* __restrict__ hs,
    const float* __restrict__ Wtag, const float* __restrict__ btag,
    float* __restrict__ out)
{
  const int m0 = blockIdx.x * 64;
  __shared__ unsigned short Bs[64 * 40];
  const int tid = threadIdx.x;
  const int lane = tid & 31;
  const int wave = tid >> 5;

  v8f acc[4] = {};

  for (int kk = 0; kk < 2 * HH; kk += 32) {
#pragma unroll
    for (int i = 0; i < 4; ++i) {
      int idx = i * 128 + tid;
      int r = idx >> 3;
      int c4 = (idx & 7) * 4;
      uint2 pb;
      if (r < KK) {
        float4 vb = *(const float4*)&Wtag[(size_t)r * (2 * HH) + kk + c4];
        pb.x = pack_bf16(vb.x, vb.y);
        pb.y = pack_bf16(vb.z, vb.w);
      } else {
        pb.x = 0u; pb.y = 0u;
      }
      *(uint2*)&Bs[r * 40 + c4] = pb;
    }
    __syncthreads();

    v16bf a = ldA(hs, 2 * HH, m0 + wave * 16, kk, lane);  // global bf16 A
#pragma unroll
    for (int ns = 0; ns < 4; ++ns) {
      v16bf bq = ldB(Bs, 40, ns * 16, 0, lane);
      acc[ns] = WMMA_BF16(a, bq, acc[ns]);
    }
    __syncthreads();
  }

#pragma unroll
  for (int ns = 0; ns < 4; ++ns) {
    int col = ns * 16 + (lane & 15);
    if (col < KK) {
      float bias = btag[col];
#pragma unroll
      for (int r = 0; r < 8; ++r) {
        int row = m0 + wave * 16 + r + ((lane < 16) ? 0 : 8);
        out[(size_t)row * KK + col] = acc[ns][r] + bias;
      }
    }
  }
}

// ---------------------------------------------------------------------------
// K4: Viterbi decode.  One block per batch row; thread j owns state j.
// Transitions cached in LDS; backpointers in ws (uint8); serial backtrace.
// ---------------------------------------------------------------------------
__global__ void __launch_bounds__(64) k_viterbi(
    const float* __restrict__ outs, const float* __restrict__ trans,
    const int* __restrict__ lens, unsigned char* __restrict__ bps,
    float* __restrict__ tagout)
{
  const int b = blockIdx.x;
  const int j = threadIdx.x;
  __shared__ float tr[KK * 53];
  __shared__ float part[64];
  __shared__ float fin[64];

  for (int i = j; i < KK * KK; i += 64) tr[(i / KK) * 53 + (i % KK)] = trans[i];
  __syncthreads();

  const int len = lens[b];
  const float* e = outs + (size_t)b * TT * KK;

  float pj = 0.0f;
  if (j < KK) pj = e[j] + tr[START_TAG * 53 + j];
  part[j] = pj;
  fin[j] = -3.0e38f;
  __syncthreads();

  for (int t = 1; t < TT; ++t) {
    float best = -3.0e38f;
    int arg = 0;
    if (j < KK) {
      float et = e[(size_t)t * KK + j];
      for (int i = 0; i < KK; ++i) {
        float v = part[i] + tr[i * 53 + j] + et;
        if (v > best) { best = v; arg = i; }   // first-max, matches argmax
      }
    }
    bool m = t < len;
    float np = pj; int bp = j;
    if (j < KK) { np = m ? best : pj; bp = m ? arg : j; }
    __syncthreads();   // all part[] reads done
    if (j < KK) {
      part[j] = np;
      bps[((size_t)b * TT + (t - 1)) * KK + j] = (unsigned char)bp;
    }
    pj = np;
    __syncthreads();
  }

  if (j < KK) fin[j] = pj + tr[j * 53 + STOP_TAG];
  __syncthreads();
  if (j == 0) {
    float best = -3.0e38f; int last = 0;
    for (int i = 0; i < KK; ++i)
      if (fin[i] > best) { best = fin[i]; last = i; }
    int ptr = last;
    for (int t = TT - 1; t >= 1; --t) {
      tagout[(size_t)b * TT + t] = (t < len) ? (float)ptr : 0.0f;
      ptr = bps[((size_t)b * TT + (t - 1)) * KK + ptr];
    }
    tagout[(size_t)b * TT + 0] = (float)ptr;   // len >= 1 always
  }
}

// ---------------------------------------------------------------------------
extern "C" void kernel_launch(void* const* d_in, const int* in_sizes, int n_in,
                              void* d_out, int out_size, void* d_ws, size_t ws_size,
                              hipStream_t stream)
{
  (void)in_sizes; (void)n_in; (void)out_size; (void)ws_size;
  const float* x     = (const float*)d_in[0];
  const int*   lens  = (const int*)d_in[1];
  /* d_in[2] = mask (derived from lens; unused) */
  const float* Wih_f = (const float*)d_in[3];
  const float* Whh_f = (const float*)d_in[4];
  const float* bih_f = (const float*)d_in[5];
  const float* bhh_f = (const float*)d_in[6];
  const float* Wih_b = (const float*)d_in[7];
  const float* Whh_b = (const float*)d_in[8];
  const float* bih_b = (const float*)d_in[9];
  const float* bhh_b = (const float*)d_in[10];
  const float* W_tag = (const float*)d_in[11];
  const float* b_tag = (const float*)d_in[12];
  const float* trans = (const float*)d_in[13];

  char* ws = (char*)d_ws;
  size_t off = 0;
  float* xgf = (float*)(ws + off);              off += (size_t)BB * TT * G4 * 4;      // 256 MB
  float* xgb = (float*)(ws + off);              off += (size_t)BB * TT * G4 * 4;      // 256 MB
  unsigned short* hsbf = (unsigned short*)(ws + off); off += (size_t)BB * TT * 2 * HH * 2; // 64 MB
  unsigned short* whf  = (unsigned short*)(ws + off); off += (size_t)G4 * HH * 2;     // 512 KB
  unsigned short* whb  = (unsigned short*)(ws + off); off += (size_t)G4 * HH * 2;     // 512 KB
  unsigned char*  bps  = (unsigned char*)(ws + off);  off += (size_t)BB * TT * KK;    // 3.4 MB

  float* outs = (float*)d_out;
  float* tago = outs + (size_t)BB * TT * KK;

  k_cvt_bf16<<<dim3((G4 * HH + 255) / 256), 256, 0, stream>>>(Whh_f, whf, G4 * HH);
  k_cvt_bf16<<<dim3((G4 * HH + 255) / 256), 256, 0, stream>>>(Whh_b, whb, G4 * HH);
  k_xg_gemm<<<dim3(BB * TT / 64, G4 / 64, 2), 128, 0, stream>>>(
      x, Wih_f, bih_f, bhh_f, Wih_b, bih_b, bhh_b, xgf, xgb);
  k_lstm<<<dim3(2), 512, 0, stream>>>(xgf, xgb, whf, whb, lens, hsbf);
  k_emit<<<dim3(BB * TT / 64), 128, 0, stream>>>(hsbf, W_tag, b_tag, outs);
  k_viterbi<<<dim3(BB), 64, 0, stream>>>(outs, trans, lens, bps, tago);
}
